// CommunicationHead_19473381720779
// MI455X (gfx1250) — matrix operand order repaired and checked by reading, not compiled, not run on hardware
//
#include <hip/hip_runtime.h>

// ---------------------------------------------------------------------------
// Problem constants (from reference)
// ---------------------------------------------------------------------------
#define INPUT_DIM 512
#define EMBED     64
#define HIDDEN    256
#define BATCH     32
#define TSTEPS    64
#define ROWS      (BATCH * TSTEPS)   // 2048
#define G3        (3 * HIDDEN)       // 768

typedef __attribute__((ext_vector_type(16))) _Float16 v16h;
typedef __attribute__((ext_vector_type(8)))  float    v8f;
typedef __attribute__((ext_vector_type(4)))  unsigned tdm_g0_t;  // D# group 0
typedef __attribute__((ext_vector_type(8)))  unsigned tdm_g1_t;  // D# group 1

union Frag {
    uint4 u[2];
    v16h  h;
};

// Load a 16-half WMMA fragment as two 16-byte vectors.
// A- and B-matrix fragments: hi half is +16 halves away.
__device__ __forceinline__ v16h load_frag(const _Float16* p, int hiOff) {
    Frag f;
    f.u[0] = *reinterpret_cast<const uint4*>(p);
    f.u[1] = *reinterpret_cast<const uint4*>(p + hiOff);
    return f.h;
}

__device__ __forceinline__ v8f wmma_f16(v16h a, v16h b, v8f c) {
    return __builtin_amdgcn_wmma_f32_16x16x32_f16(false, a, false, b,
                                                  (short)0, c, false, false);
}

__device__ __forceinline__ float sigmoidf_fast(float x) {
    return 1.0f / (1.0f + __expf(-x));
}

// ---------------------------------------------------------------------------
// Tensor Data Mover: DMA a 2D tile of f16 (rows x elemsPerRow) from global
// memory into LDS. D# layout per CDNA5 ISA (CDNA5 ISA):
//   group0: count=1, lds_addr, global_addr, type=2
//   group1: data_size=2B, tensor_dim0/1 (OOB -> zero fill), tile_dim0/1,
//           tensor_dim0_stride (row pitch in elements)
// tensor_dim0/1 are expressed as extent remaining from the tile start, so
// rows/cols beyond the tensor are written to LDS as zeros (edge handling).
// Completion tracked with TENSORcnt.
// ---------------------------------------------------------------------------
__device__ __forceinline__ void tdm_load_tile_f16(
        unsigned ldsByteOff, const _Float16* gptr,
        unsigned tileElemsPerRow, unsigned tileRows,
        unsigned tensorElems0, unsigned tensorRows1,
        unsigned long long rowStrideElems) {
    unsigned long long ga = (unsigned long long)(uintptr_t)gptr;
    tdm_g0_t g0;
    g0[0] = 1u;                                              // count = 1
    g0[1] = ldsByteOff;                                      // lds_addr
    g0[2] = (unsigned)ga;                                    // global_addr[31:0]
    g0[3] = (unsigned)((ga >> 32) & 0x01FFFFFFu) | (2u << 30); // addr[56:32] | type=2
    tdm_g1_t g1;
    g1[0] = 0x00010000u;                                     // data_size = 2 bytes
    g1[1] = (tensorElems0 & 0xFFFFu) << 16;                  // tensor_dim0[15:0]
    g1[2] = (tensorElems0 >> 16) | ((tensorRows1 & 0xFFFFu) << 16); // dim0 hi | dim1 lo
    g1[3] = (tensorRows1 >> 16) | (tileElemsPerRow << 16);   // dim1 hi | tile_dim0
    g1[4] = tileRows;                                        // tile_dim1 (tile_dim2=0)
    g1[5] = (unsigned)rowStrideElems;                        // tensor_dim0_stride lo
    g1[6] = (unsigned)(rowStrideElems >> 32);                // stride hi | dim1_stride lo
    g1[7] = 0u;
    asm volatile("tensor_load_to_lds %0, %1" :: "s"(g0), "s"(g1) : "memory");
}

// ---------------------------------------------------------------------------
// fp32 -> f16 conversion
// ---------------------------------------------------------------------------
__global__ void cvt_f32_to_f16(const float* __restrict__ src,
                               _Float16* __restrict__ dst, int n) {
    int i = blockIdx.x * blockDim.x + threadIdx.x;
    if (i < n) dst[i] = (_Float16)src[i];
}

// ---------------------------------------------------------------------------
// Embedding gather: X[(b*T+t)*E + e] = f16(embed[target[b*T+t]*E + e])
// ---------------------------------------------------------------------------
__global__ void gather_embed(const int* __restrict__ target,
                             const float* __restrict__ embed,
                             _Float16* __restrict__ X, int n) {
    int i = blockIdx.x * blockDim.x + threadIdx.x;
    if (i < n) {
        int row = i >> 6;          // EMBED == 64
        int e   = i & 63;
        int tok = target[row];
        X[i] = (_Float16)embed[tok * EMBED + e];
    }
}

// ---------------------------------------------------------------------------
// Generic tiled WMMA GEMM:  C(MxN,f32,row-major) = A(MxK,f16) * Bm(NxK,f16)^T + bias
// Block tile 128x128, 512 threads = 16 waves, each wave computes 16x64.
// Panels are staged into LDS by the Tensor Data Mover (wave 0 issues two
// tensor_load_to_lds, waits TENSORcnt==0, then a workgroup barrier publishes
// the panel). K chunked at KC = min(K,256) (<=128 KB LDS).
// ---------------------------------------------------------------------------
__global__ __launch_bounds__(512)
void gemm_wmma_f16(const _Float16* __restrict__ A,
                   const _Float16* __restrict__ Bm,
                   const float* __restrict__ bias,
                   float* __restrict__ C,
                   int M, int N, int K) {
    extern __shared__ _Float16 smem[];
    const int KC = (K <= 256) ? K : 256;
    _Float16* As = smem;             // 128 * KC halves
    _Float16* Bs = smem + 128 * KC;  // 128 * KC halves

    const int tid   = threadIdx.x;
    const int mBase = blockIdx.y * 128;
    const int nBase = blockIdx.x * 128;

    const int wave  = tid >> 5;
    const int lane  = tid & 31;
    const int lmod  = lane & 15;
    const int khalf = lane >> 4;
    const int mt    = wave >> 1;        // 0..7 -> 16-row strip
    const int ntq   = (wave & 1) * 4;   // 0 or 4 -> 4 N-tiles of 16

    const unsigned asOff = (unsigned)(uintptr_t)As;
    const unsigned bsOff = (unsigned)(uintptr_t)Bs;

    v8f acc[4] = {};

    for (int k0 = 0; k0 < K; k0 += KC) {
        // ---- TDM stage: global -> LDS, OOB rows zero-filled by tensor_dim1 ----
        if (wave == 0) {
            tdm_load_tile_f16(asOff, A + (size_t)mBase * K + k0,
                              (unsigned)KC, 128u,
                              (unsigned)(K - k0), (unsigned)(M - mBase),
                              (unsigned long long)K);
            tdm_load_tile_f16(bsOff, Bm + (size_t)nBase * K + k0,
                              (unsigned)KC, 128u,
                              (unsigned)(K - k0), (unsigned)(N - nBase),
                              (unsigned long long)K);
            __builtin_amdgcn_s_wait_tensorcnt(0);
        }
        __syncthreads();

        // ---- WMMA over this K chunk ----
        for (int ks = 0; ks < KC; ks += 32) {
            v16h a = load_frag(As + (mt * 16 + lmod) * KC + ks + khalf * 8, 16);
#pragma unroll
            for (int j = 0; j < 4; ++j) {
                v16h b = load_frag(Bs + (ntq + j) * 16 * KC + lmod * KC + ks + khalf * 8, 16);
                acc[j] = wmma_f16(a, b, acc[j]);
            }
        }
        __syncthreads();
    }

    // ---- epilogue: bias + guarded store ----
#pragma unroll
    for (int j = 0; j < 4; ++j) {
        int col = nBase + (ntq + j) * 16 + lmod;
        bool cv = (col < N);
        float bv = cv ? bias[col] : 0.0f;
#pragma unroll
        for (int r = 0; r < 8; ++r) {
            int row = mBase + mt * 16 + r + 8 * khalf;
            if (cv && row < M)
                C[(size_t)row * N + col] = acc[j][r] + bv;
        }
    }
}

// ---------------------------------------------------------------------------
// GRU recurrence: one persistent workgroup (512 threads / 16 waves).
// Per step: gh = h @ W_hh^T  via WMMA (2 M-tiles x 48 N-tiles x K=256).
// Wave w owns columns j = w*16 + (lane&15); its 6 accumulator tiles are the
// (r,z,n) gate tiles for exactly those columns -> elementwise GRU stays in
// registers, h kept in per-lane fp32 regs + f16 LDS for the WMMA A operand.
// ---------------------------------------------------------------------------
__global__ __launch_bounds__(512)
void gru_recurrence_wmma(const float* __restrict__ GI,      // ROWS x 768 (has b_ih)
                         const _Float16* __restrict__ Whh,  // 768 x 256 f16
                         const float* __restrict__ b_hh,    // 768
                         const float* __restrict__ H0,      // 32 x 256 f32
                         _Float16* __restrict__ Hout) {     // ROWS x 256 f16
    __shared__ _Float16 hsh[BATCH * HIDDEN];  // 16 KB

    const int tid   = threadIdx.x;
    const int wave  = tid >> 5;
    const int lane  = tid & 31;
    const int lmod  = lane & 15;
    const int khalf = lane >> 4;
    const int j     = wave * 16 + lmod;       // hidden column 0..255

    float bh[3];
#pragma unroll
    for (int g = 0; g < 3; ++g) bh[g] = b_hh[g * HIDDEN + j];

    float hreg[2][8];
#pragma unroll
    for (int mt = 0; mt < 2; ++mt)
#pragma unroll
        for (int r = 0; r < 8; ++r) {
            int b = mt * 16 + r + 8 * khalf;
            float h = H0[b * HIDDEN + j];
            hreg[mt][r] = h;
            hsh[b * HIDDEN + j] = (_Float16)h;
        }
    __syncthreads();

    for (int t = 0; t < TSTEPS; ++t) {
        v8f cc[2][3] = {};
        for (int ks = 0; ks < HIDDEN; ks += 32) {
            v16h a0 = load_frag(hsh + (lmod)      * HIDDEN + ks + khalf * 8, 16);
            v16h a1 = load_frag(hsh + (16 + lmod) * HIDDEN + ks + khalf * 8, 16);
#pragma unroll
            for (int g = 0; g < 3; ++g) {
                const _Float16* bp =
                    Whh + (size_t)(g * HIDDEN + wave * 16 + lmod) * HIDDEN + ks + khalf * 8;
                v16h b = load_frag(bp, 16);
                cc[0][g] = wmma_f16(a0, b, cc[0][g]);
                cc[1][g] = wmma_f16(a1, b, cc[1][g]);
            }
        }
        __syncthreads();  // all waves done reading hsh

#pragma unroll
        for (int mt = 0; mt < 2; ++mt) {
#pragma unroll
            for (int r = 0; r < 8; ++r) {
                int b   = mt * 16 + r + 8 * khalf;
                int row = b * TSTEPS + t;
                const float* gi = GI + (size_t)row * G3;
                float hr = cc[mt][0][r] + bh[0];
                float hz = cc[mt][1][r] + bh[1];
                float hn = cc[mt][2][r] + bh[2];
                float rg = sigmoidf_fast(gi[j] + hr);
                float zg = sigmoidf_fast(gi[HIDDEN + j] + hz);
                float ng = tanhf(gi[2 * HIDDEN + j] + rg * hn);
                float h  = (1.0f - zg) * ng + zg * hreg[mt][r];
                hreg[mt][r] = h;
                hsh[b * HIDDEN + j] = (_Float16)h;
                Hout[(size_t)row * HIDDEN + j] = (_Float16)h;
            }
        }
        __syncthreads();  // hsh ready for next step
    }
}

// ---------------------------------------------------------------------------
// Host side
// ---------------------------------------------------------------------------
extern "C" void kernel_launch(void* const* d_in, const int* in_sizes, int n_in,
                              void* d_out, int out_size, void* d_ws, size_t ws_size,
                              hipStream_t stream) {
    (void)n_in; (void)out_size; (void)ws_size;

    const float* state  = (const float*)d_in[0];
    const int*   target = (const int*)  d_in[1];
    const float* embed  = (const float*)d_in[2];
    const float* Wp     = (const float*)d_in[3];
    const float* bp     = (const float*)d_in[4];
    const float* W_ih   = (const float*)d_in[5];
    const float* W_hh   = (const float*)d_in[6];
    const float* b_ih   = (const float*)d_in[7];
    const float* b_hh   = (const float*)d_in[8];
    const float* Wo     = (const float*)d_in[9];
    const float* bo     = (const float*)d_in[10];
    const int V = in_sizes[10];  // vocab size from bo length

    // workspace layout (256B aligned slices)
    char* ws = (char*)d_ws;
    size_t off = 0;
    auto take = [&](size_t bytes) {
        size_t o = off;
        off += (bytes + 255) & ~(size_t)255;
        return o;
    };
    _Float16* Xf    = (_Float16*)(ws + take((size_t)ROWS * EMBED * 2));
    _Float16* Hf    = (_Float16*)(ws + take((size_t)ROWS * HIDDEN * 2));
    _Float16* st16  = (_Float16*)(ws + take((size_t)BATCH * INPUT_DIM * 2));
    _Float16* Wp16  = (_Float16*)(ws + take((size_t)HIDDEN * INPUT_DIM * 2));
    _Float16* Wih16 = (_Float16*)(ws + take((size_t)G3 * EMBED * 2));
    _Float16* Whh16 = (_Float16*)(ws + take((size_t)G3 * HIDDEN * 2));
    _Float16* Wo16  = (_Float16*)(ws + take((size_t)V * HIDDEN * 2));
    float*    GIb   = (float*)   (ws + take((size_t)ROWS * G3 * 4));
    float*    H0b   = (float*)   (ws + take((size_t)BATCH * HIDDEN * 4));

    auto cvt = [&](const float* s, _Float16* d, int n) {
        cvt_f32_to_f16<<<(n + 255) / 256, 256, 0, stream>>>(s, d, n);
    };
    cvt(state, st16, BATCH * INPUT_DIM);
    cvt(Wp,    Wp16, HIDDEN * INPUT_DIM);
    cvt(W_ih,  Wih16, G3 * EMBED);
    cvt(W_hh,  Whh16, G3 * HIDDEN);
    cvt(Wo,    Wo16, V * HIDDEN);

    {
        int n = ROWS * EMBED;
        gather_embed<<<(n + 255) / 256, 256, 0, stream>>>(target, embed, Xf, n);
    }

    auto shbytes = [](int K) { return (size_t)512 * (size_t)((K <= 256) ? K : 256); };

    // h0 = state @ Wp^T + bp : M=32, N=256, K=512
    {
        dim3 grid((HIDDEN + 127) / 128, (BATCH + 127) / 128);
        gemm_wmma_f16<<<grid, 512, shbytes(INPUT_DIM), stream>>>(
            st16, Wp16, bp, H0b, BATCH, HIDDEN, INPUT_DIM);
    }
    // GI = X @ W_ih^T + b_ih : M=2048, N=768, K=64
    {
        dim3 grid((G3 + 127) / 128, (ROWS + 127) / 128);
        gemm_wmma_f16<<<grid, 512, shbytes(EMBED), stream>>>(
            Xf, Wih16, b_ih, GIb, ROWS, G3, EMBED);
    }
    // GRU recurrence (single persistent workgroup)
    gru_recurrence_wmma<<<1, 512, 0, stream>>>(GIb, Whh16, b_hh, H0b, Hf);

    // logits = H @ Wo^T + bo : M=2048, N=V, K=256  -> d_out (B,T,V)
    {
        dim3 grid((V + 127) / 128, (ROWS + 127) / 128);
        gemm_wmma_f16<<<grid, 512, shbytes(HIDDEN), stream>>>(
            Hf, Wo16, bo, (float*)d_out, ROWS, V, HIDDEN);
    }
}
